// FPMAApproxBF16_43602507989109
// MI455X (gfx1250) — matrix operand order repaired and checked by reading, not compiled, not run on hardware
//
#include <hip/hip_runtime.h>
#include <math.h>

// ---------------------------------------------------------------------------
// FPMA-approx bf16 GEMM -> CDNA5 WMMA. Memory-bound: ~65MB traffic (C in 32MB
// + D out 32MB dominate) @ 23.3TB/s ~= 2.8us; 2.1 GFLOP of bf16 WMMA is
// negligible. So: route the multiply through v_wmma_f32_16x16x32_bf16, make
// ALL global traffic 128-bit coalesced, and use the gfx1250 async-to-LDS path
// for the A tile copy.
//   D = C + A*B, A:[M,64] bf16 row-major, B:[64,N] bf16, C/D:[M,N] bf16.
// ---------------------------------------------------------------------------

typedef __attribute__((ext_vector_type(16))) __bf16    v16bf;
typedef __attribute__((ext_vector_type(8)))  float     v8f;
typedef __attribute__((ext_vector_type(4)))  unsigned  u32x4;
typedef int i32x4v __attribute__((vector_size(16)));   // matches builtin param

typedef const __attribute__((address_space(1))) void* gvoid_p;
typedef       __attribute__((address_space(3))) void* lvoid_p;
typedef __attribute__((address_space(1))) i32x4v* gvec_p;
typedef __attribute__((address_space(3))) i32x4v* lvec_p;

#define KD   64     // K (fixed by reference)
#define BM   64     // block tile M (4 wave-tiles)
#define BN   32     // block tile N (2 wave-tiles)
#define LDA  72     // LDS elem stride, A tile rows   (mult of 8, conflict pad)
#define LDB  72     // LDS elem stride, Bt rows
#define LDD  36     // LDS dword stride, f32 D tile   (mult of 4, conflict pad)

#if __has_builtin(__builtin_amdgcn_global_load_async_to_lds_b128)
#define HAVE_ASYNC_LDS 1
#endif

__device__ __forceinline__ unsigned short f2bf(float f) {
    unsigned u = __float_as_uint(f);
    unsigned r = (u + 0x7FFFu + ((u >> 16) & 1u)) >> 16;   // round-nearest-even
    return (unsigned short)r;
}
__device__ __forceinline__ float bf2f(unsigned short h) {
    return __uint_as_float(((unsigned)h) << 16);
}

union Frag  { u32x4 q[2]; v16bf v; };
union Pack8 { u32x4 q; unsigned short s[8]; };
union F8    { u32x4 q[2]; float f[8]; };

__global__ __launch_bounds__(256)
void fpma_wmma_bf16_kernel(const unsigned short* __restrict__ A,
                           const unsigned short* __restrict__ B,
                           const unsigned short* __restrict__ C,
                           unsigned short* __restrict__ Out,
                           int M, int N) {
    __shared__ unsigned short As[BM * LDA];   // A tile, [m][k], k contiguous
    __shared__ unsigned short Bt[BN * LDB];   // B tile transposed, [n][k]
    __shared__ float          Dt[BM * LDD];   // f32 result tile, [m][n]

    const int t  = threadIdx.x;
    const int m0 = blockIdx.y * BM;
    const int n0 = blockIdx.x * BN;

    // ---- A tile: BM x 64 bf16 = 8KB. Async DMA global->LDS, 2 x 16B/thread -
    #pragma unroll
    for (int it = 0; it < 2; ++it) {
        int idx = t + it * 256;                    // 512 chunks of 8 elements
        int row = idx >> 3;                        // 8 chunks per 64-elem row
        int c8  = (idx & 7) * 8;
        const unsigned short* gsrc = A + (size_t)(m0 + row) * KD + c8;
        unsigned short*       ldst = As + row * LDA + c8;
#if defined(HAVE_ASYNC_LDS)
        gvoid_p g1 = (gvoid_p)gsrc;                // generic -> AS1 (proven ok)
        lvoid_p l3 = (lvoid_p)ldst;                // generic -> AS3
        __builtin_amdgcn_global_load_async_to_lds_b128(
            (gvec_p)g1, (lvec_p)l3, 0, 0);
#else
        *(u32x4*)ldst = *(const u32x4*)gsrc;
#endif
    }
    // ---- B tile: 64 x BN bf16 = 4KB, coalesced b128 load, transposed store -
    {
        int k  = t >> 2;                           // 64 rows
        int c8 = (t & 3) * 8;                      // 4 chunks of 8 cols
        u32x4 d = *(const u32x4*)(B + (size_t)k * N + n0 + c8);
        const unsigned short* ds = (const unsigned short*)&d;
        #pragma unroll
        for (int j = 0; j < 8; ++j)
            Bt[(c8 + j) * LDB + k] = ds[j];
    }
#if defined(HAVE_ASYNC_LDS)
#if __has_builtin(__builtin_amdgcn_s_wait_asynccnt)
    __builtin_amdgcn_s_wait_asynccnt(0);
#else
    asm volatile("s_wait_asynccnt 0x0" ::: "memory");
#endif
#endif
    __syncthreads();

    const int wave = t >> 5;                       // 8 waves / block
    const int lane = t & 31;
    const int half = lane >> 4;
    const int r    = lane & 15;
    const int wm   = (wave & 3) * 16;              // wave tile origin (local)
    const int wn   = (wave >> 2) * 16;

    // ---- WMMA fragments from LDS (ds_load_b128) ----------------------------
    // A (16x32): elems 0..7 = K[half*8..+8), elems 8..15 = K[16+half*8..+8)
    Frag a0, a1, b0, b1;
    {
        const unsigned short* ar = As + (wm + r) * LDA;
        a0.q[0] = *(const u32x4*)(ar +  0 + half * 8);
        a0.q[1] = *(const u32x4*)(ar + 16 + half * 8);
        a1.q[0] = *(const u32x4*)(ar + 32 + half * 8);
        a1.q[1] = *(const u32x4*)(ar + 48 + half * 8);
        // B (32x16): lane = column, elems 0..7 = K[half*8..+8), elems 8..15 = K[16+half*8..+8)
        const unsigned short* br = Bt + (wn + r) * LDB;
        b0.q[0] = *(const u32x4*)(br +  0 + half * 8);
        b0.q[1] = *(const u32x4*)(br + 16 + half * 8);
        b1.q[0] = *(const u32x4*)(br + 32 + half * 8);
        b1.q[1] = *(const u32x4*)(br + 48 + half * 8);
    }

    // ---- acc = A*B over K=64 (two K32 WMMAs), zero C operand ---------------
    v8f acc = {0.f, 0.f, 0.f, 0.f, 0.f, 0.f, 0.f, 0.f};
    acc = __builtin_amdgcn_wmma_f32_16x16x32_bf16(false, a0.v, false, b0.v,
                                                  (short)0, acc, false, false);
    acc = __builtin_amdgcn_wmma_f32_16x16x32_bf16(false, a1.v, false, b1.v,
                                                  (short)0, acc, false, false);

    // ---- linearize fragment through LDS: VGPR i = row half*8+i, lane = col -
    {
        float* dr = Dt + (wm + half * 8) * LDD + wn + r;
        #pragma unroll
        for (int i = 0; i < 8; ++i)
            dr[i * LDD] = acc[i];
    }
    __syncthreads();

    // ---- +C and store, fully coalesced 16B/thread --------------------------
    {
        int row = t >> 2;                          // 64 rows
        int c8  = (t & 3) * 8;                     // 4 chunks of 8 cols
        F8 d;
        d.q[0] = *(const u32x4*)(Dt + row * LDD + c8);
        d.q[1] = *(const u32x4*)(Dt + row * LDD + c8 + 4);

        Pack8 c;
        c.q = *(const u32x4*)(C + (size_t)(m0 + row) * N + n0 + c8);

        Pack8 o;
        #pragma unroll
        for (int j = 0; j < 8; ++j)
            o.s[j] = f2bf(d.f[j] + bf2f(c.s[j]));

        *(u32x4*)(Out + (size_t)(m0 + row) * N + n0 + c8) = o.q;
    }
}

extern "C" void kernel_launch(void* const* d_in, const int* in_sizes, int n_in,
                              void* d_out, int out_size, void* d_ws, size_t ws_size,
                              hipStream_t stream) {
    (void)n_in; (void)d_ws; (void)ws_size; (void)out_size;

    const unsigned short* A = (const unsigned short*)d_in[0];
    const unsigned short* B = (const unsigned short*)d_in[1];
    const unsigned short* C = (const unsigned short*)d_in[2];
    unsigned short* Out = (unsigned short*)d_out;

    // Recover dims host-side from flat sizes: s0=M*K, s1=K*N, s2=M*N
    const double s0 = (double)in_sizes[0];
    const double s1 = (double)in_sizes[1];
    const double s2 = (double)in_sizes[2];
    int K = (int)(sqrt(s0 * s1 / s2) + 0.5);       // == 64 for this reference
    if (K <= 0) K = KD;
    int M = (int)(s0 / K + 0.5);
    int N = (int)(s1 / K + 0.5);

    dim3 grid((unsigned)(N / BN), (unsigned)(M / BM));
    dim3 block(256);
    fpma_wmma_bf16_kernel<<<grid, block, 0, stream>>>(A, B, C, Out, M, N);
}
